// Attention_7868380086810
// MI455X (gfx1250) — matrix operand order repaired and checked by reading, not compiled, not run on hardware
//
#include <hip/hip_runtime.h>
#include <hip/hip_fp16.h>

typedef __attribute__((ext_vector_type(16))) _Float16 v16h;
typedef __attribute__((ext_vector_type(8)))  _Float16 v8h;
typedef __attribute__((ext_vector_type(4)))  _Float16 v4h;
typedef __attribute__((ext_vector_type(8)))  float    v8f;
typedef __attribute__((ext_vector_type(4)))  float    v4f;

#define B_    2
#define S_    2048
#define D_    1024
#define H_    16
#define HD_   64
#define BH_   (B_*H_)      // 32
#define MTOT  (B_*S_)      // 4096
#define NEG_INF (-__builtin_inff())

static __device__ __forceinline__ v16h mk16(v8h lo, v8h hi) {
  return __builtin_shufflevector(lo, hi, 0,1,2,3,4,5,6,7,8,9,10,11,12,13,14,15);
}

static __device__ __forceinline__ v8f wmma_f16(v16h a, v16h b, v8f c) {
  // D = A(16x32 f16) * B(32x16 f16) + C(16x16 f32)
  return __builtin_amdgcn_wmma_f32_16x16x32_f16(
      /*neg_a=*/false, a, /*neg_b=*/false, b,
      /*c_mod=*/(short)0, c, /*reuse_a=*/false, /*reuse_b=*/false);
}

// A fragment: 16(M) x 32(K), rows K-contiguous at p + m*ld.
// Lane l holds row m=l&15; VGPR0-3 = K[kb..kb+7], VGPR4-7 = K[kb+16..kb+23], kb=(l>=16)?8:0.
static __device__ __forceinline__ v16h ldA(const _Float16* p, size_t ld, int lane) {
  int m  = lane & 15;
  int kb = (lane >> 4) * 8;
  const _Float16* r = p + (size_t)m * ld + kb;
  return mk16(*(const v8h*)r, *(const v8h*)(r + 16));
}

// B fragment: 32(K) x 16(N) from W[n,k] with k contiguous at p + n*ld.
// Lane l holds col n=l&15; VGPR0-3 = K[kb..kb+7], VGPR4-7 = K[kb+16..kb+23], kb=(l>=16)?8:0.
static __device__ __forceinline__ v16h ldB(const _Float16* p, size_t ld, int lane) {
  int n   = lane & 15;
  int kb2 = (lane >> 4) * 8;
  const _Float16* r = p + (size_t)n * ld + kb2;
  return mk16(*(const v8h*)r, *(const v8h*)(r + 16));
}

static __device__ __forceinline__ float half16_max(float v) {
  #pragma unroll
  for (int m = 1; m < 16; m <<= 1) v = fmaxf(v, __shfl_xor(v, m, 32));
  return v;
}
static __device__ __forceinline__ float half16_sum(float v) {
  #pragma unroll
  for (int m = 1; m < 16; m <<= 1) v += __shfl_xor(v, m, 32);
  return v;
}

// ---------------------------------------------------------------- convert (vectorized x4)
__global__ void cvt_f32_f16(const float* __restrict__ src,
                            _Float16* __restrict__ dst, int n4) {
  int i = blockIdx.x * blockDim.x + threadIdx.x;
  int stride = gridDim.x * blockDim.x;
  for (; i < n4; i += stride) {
    v4f s = ((const v4f*)src)[i];
    v4h d;
    #pragma unroll
    for (int j = 0; j < 4; ++j) d[j] = (_Float16)s[j];
    ((v4h*)dst)[i] = d;
  }
}

// Rolled, manually 2x-unrolled ping-pong NT GEMM mainloop: 16x64 tile.
// Buffers are rewritten by loads each iteration (no register copies), and one
// full stage of loads stays in flight behind every WMMA group.
static __device__ __forceinline__ void gemm_mainloop(
    const _Float16* __restrict__ Ap, const _Float16* __restrict__ Bp,
    int lane, v8f acc[4]) {
  v16h a0, a1, b0[4], b1[4];
  a0 = ldA(Ap, D_, lane);
  #pragma unroll
  for (int c = 0; c < 4; ++c) b0[c] = ldB(Bp + (size_t)c * 16 * D_, D_, lane);

  int k = 0;
  #pragma clang loop unroll(disable)
  for (; k + 64 < D_; k += 64) {
    a1 = ldA(Ap + k + 32, D_, lane);                       // prefetch stage k+32
    #pragma unroll
    for (int c = 0; c < 4; ++c)
      b1[c] = ldB(Bp + (size_t)c * 16 * D_ + k + 32, D_, lane);
    #pragma unroll
    for (int c = 0; c < 4; ++c) acc[c] = wmma_f16(a0, b0[c], acc[c]);   // stage k

    a0 = ldA(Ap + k + 64, D_, lane);                       // prefetch stage k+64
    #pragma unroll
    for (int c = 0; c < 4; ++c)
      b0[c] = ldB(Bp + (size_t)c * 16 * D_ + k + 64, D_, lane);
    #pragma unroll
    for (int c = 0; c < 4; ++c) acc[c] = wmma_f16(a1, b1[c], acc[c]);   // stage k+32
  }
  // tail: k == D_-64 here; stages k and k+32 remain, buffer0 holds stage k.
  a1 = ldA(Ap + k + 32, D_, lane);
  #pragma unroll
  for (int c = 0; c < 4; ++c)
    b1[c] = ldB(Bp + (size_t)c * 16 * D_ + k + 32, D_, lane);
  #pragma unroll
  for (int c = 0; c < 4; ++c) acc[c] = wmma_f16(a0, b0[c], acc[c]);
  #pragma unroll
  for (int c = 0; c < 4; ++c) acc[c] = wmma_f16(a1, b1[c], acc[c]);
}

// ---------------------------------------------------------------- QKV GEMM
// C[M=4096, E=3072] = Xf16[4096,1024] * Wqkv[3072,1024]^T
// Each wave: 16x64 output tile; 64-wide N tile == exactly one head (q/k/v).
__global__ __launch_bounds__(128) void qkv_gemm(
    const _Float16* __restrict__ X, const _Float16* __restrict__ W,
    _Float16* __restrict__ qb, _Float16* __restrict__ kb,
    _Float16* __restrict__ vt) {
  const int lane = threadIdx.x & 31, wv = threadIdx.x >> 5;
  const int flat = blockIdx.x * 4 + wv;       // 12288 wave tiles
  const int nt = flat % 48;                   // head3 index (0..47)
  const int mt = flat / 48;                   // 0..255

  v8f acc[4] = {};
  gemm_mainloop(X + (size_t)mt * 16 * D_, W + (size_t)nt * 64 * D_, lane, acc);

  const int n = lane & 15, base8 = (lane >> 4) * 8;
  const int bI = (mt * 16) >> 11;             // 16-row tile never crosses batch
  const int sBase = (mt * 16) & (S_ - 1);
  if (nt < 32) {                              // Q or K head -> [b,h,s,hd]
    _Float16* dst = (nt < 16) ? qb : kb;
    const int h = nt & 15;
    #pragma unroll
    for (int c = 0; c < 4; ++c)
      #pragma unroll
      for (int i = 0; i < 8; ++i) {
        int srow = sBase + base8 + i;
        dst[(((size_t)bI * H_ + h) * S_ + srow) * HD_ + c * 16 + n] =
            (_Float16)acc[c][i];
      }
  } else {                                    // V head -> transposed [b,h,hd,s]
    const int h = nt - 32;
    #pragma unroll
    for (int c = 0; c < 4; ++c)
      #pragma unroll
      for (int i = 0; i < 8; ++i) {
        int srow = sBase + base8 + i;
        int hd = c * 16 + n;
        vt[(((size_t)bI * H_ + h) * HD_ + hd) * S_ + srow] = (_Float16)acc[c][i];
      }
  }
}

// ---------------------------------------------------------------- RoPE
// qk laid out [BH, S, HD]; pairs (i, i+32); scale folds softmax 1/sqrt(HD) for Q.
__global__ void rope_kernel(_Float16* __restrict__ qk, float scale, int npairs) {
  const float lnb = 9.210340371976184f / 32.0f;   // ln(10000)/32
  int idx = blockIdx.x * blockDim.x + threadIdx.x;
  int stride = gridDim.x * blockDim.x;
  for (; idx < npairs; idx += stride) {
    int i    = idx & 31;
    int srow = (idx >> 5) & (S_ - 1);
    int rw   = idx >> 16;                         // / (32*2048)
    size_t base = ((size_t)rw * S_ + srow) * HD_;
    float x1 = (float)qk[base + i];
    float x2 = (float)qk[base + i + 32];
    float th = __expf(-(float)i * lnb);
    float f  = (float)srow * th;
    float c  = __cosf(f), sn = __sinf(f);
    qk[base + i]      = (_Float16)((x1 * c - x2 * sn) * scale);
    qk[base + i + 32] = (_Float16)((x2 * c + x1 * sn) * scale);
  }
}

// ---------------------------------------------------------------- flash attention
// One wave per 16-row q tile. K/V stream from L2 (whole head's K+V = 512KB << 192MB L2).
// V loads + next-K loads are issued before the softmax VALU stretch so they are
// in flight across the shuffle/exp section and the LDS P round-trip.
__global__ __launch_bounds__(128) void flash_attn(
    const _Float16* __restrict__ Q, const _Float16* __restrict__ K,
    const _Float16* __restrict__ VT, _Float16* __restrict__ O) {
  __shared__ _Float16 pbuf[4][16 * 40];           // per-wave P tile, padded rows
  const int lane = threadIdx.x & 31, wv = threadIdx.x >> 5;
  const int tile = blockIdx.x * 4 + wv;           // 4096 tiles
  const int st = tile & 127;                      // q tile within (b,h)
  const int bh = tile >> 7;                       // 0..31
  const int q0 = st * 16;
  const int n = lane & 15, base8 = (lane >> 4) * 8;

  const _Float16* qrow = Q + ((size_t)bh * S_ + q0) * HD_;
  v16h qa0 = ldA(qrow, HD_, lane);                // hd 0..31
  v16h qa1 = ldA(qrow + 32, HD_, lane);           // hd 32..63

  v8f acc[4] = {};
  float mrow[8], lrow[8];
  #pragma unroll
  for (int i = 0; i < 8; ++i) { mrow[i] = NEG_INF; lrow[i] = 0.f; }

  const int kend = q0 + 16;

  // preload K fragments for first key tile: [j][t] = keys j*16.., hd t*32..
  v16h kf[2][2];
  {
    const _Float16* kr = K + ((size_t)bh * S_) * HD_;
    kf[0][0] = ldB(kr, HD_, lane);
    kf[0][1] = ldB(kr + 32, HD_, lane);
    kf[1][0] = ldB(kr + 16 * HD_, HD_, lane);
    kf[1][1] = ldB(kr + 16 * HD_ + 32, HD_, lane);
  }

  for (int kt = 0; kt < kend; kt += 32) {
    // ---- scores: two 16-key subtiles
    v8f s0 = {}, s1 = {};
    s0 = wmma_f16(qa0, kf[0][0], s0);
    s0 = wmma_f16(qa1, kf[0][1], s0);
    s1 = wmma_f16(qa0, kf[1][0], s1);
    s1 = wmma_f16(qa1, kf[1][1], s1);

    // ---- issue V loads for this tile (consumed after softmax)
    v16h vb[4];
    #pragma unroll
    for (int c = 0; c < 4; ++c)
      vb[c] = ldB(VT + ((size_t)bh * HD_ + c * 16) * S_ + kt, S_, lane);

    // ---- prefetch next key tile's K fragments
    if (kt + 32 < kend) {
      const _Float16* kr = K + ((size_t)bh * S_ + kt + 32) * HD_;
      kf[0][0] = ldB(kr, HD_, lane);
      kf[0][1] = ldB(kr + 32, HD_, lane);
      kf[1][0] = ldB(kr + 16 * HD_, HD_, lane);
      kf[1][1] = ldB(kr + 16 * HD_ + 32, HD_, lane);
    }

    // ---- causal mask (only tiles touching the diagonal)
    #pragma unroll
    for (int j = 0; j < 2; ++j) {
      if (kt + j * 16 + 15 > q0) {
        int col = kt + j * 16 + n;
        #pragma unroll
        for (int i = 0; i < 8; ++i) {
          int row = q0 + base8 + i;
          if (j == 0) { if (col > row) s0[i] = NEG_INF; }
          else        { if (col > row) s1[i] = NEG_INF; }
        }
      }
    }

    // ---- online softmax (rows 0-7 in lanes 0-15, rows 8-15 in lanes 16-31)
    float alpha[8];
    #pragma unroll
    for (int i = 0; i < 8; ++i) {
      float t = half16_max(fmaxf(s0[i], s1[i]));
      float mn = fmaxf(mrow[i], t);
      alpha[i] = __expf(mrow[i] - mn);
      mrow[i] = mn;
      float p0 = __expf(s0[i] - mn);
      float p1 = __expf(s1[i] - mn);
      s0[i] = p0; s1[i] = p1;
      lrow[i] = lrow[i] * alpha[i] + half16_sum(p0 + p1);
    }
    #pragma unroll
    for (int c = 0; c < 4; ++c)
      #pragma unroll
      for (int i = 0; i < 8; ++i) acc[c][i] *= alpha[i];

    // ---- reshape P (D-layout) -> A-fragment via wave-private LDS tile
    _Float16* pb = pbuf[wv];
    #pragma unroll
    for (int i = 0; i < 8; ++i) {
      pb[(base8 + i) * 40 + n]      = (_Float16)s0[i];
      pb[(base8 + i) * 40 + 16 + n] = (_Float16)s1[i];
    }
    asm volatile("s_wait_dscnt 0x0" ::: "memory");  // same-wave LDS ordered
    __builtin_amdgcn_wave_barrier();
    v16h pa = ldA(pb, 40, lane);                    // P as 16x32 A fragment

    // ---- O += P * V
    #pragma unroll
    for (int c = 0; c < 4; ++c) acc[c] = wmma_f16(pa, vb[c], acc[c]);
  }

  // ---- epilogue: normalize and store f16 O in [b, s, d]
  const int bI = bh >> 4, h = bh & 15;
  #pragma unroll
  for (int i = 0; i < 8; ++i) {
    float r = 1.0f / lrow[i];
    int srow = q0 + base8 + i;
    #pragma unroll
    for (int c = 0; c < 4; ++c)
      O[((size_t)bI * S_ + srow) * D_ + h * HD_ + c * 16 + n] =
          (_Float16)(acc[c][i] * r);
  }
}

// ---------------------------------------------------------------- output GEMM
// out[M=4096, E=1024] = O[4096,1024] * Wout[1024,1024]^T + b_out, f32 result.
__global__ __launch_bounds__(128) void out_gemm(
    const _Float16* __restrict__ A, const _Float16* __restrict__ W,
    const float* __restrict__ bias, float* __restrict__ out) {
  const int lane = threadIdx.x & 31, wv = threadIdx.x >> 5;
  const int flat = blockIdx.x * 4 + wv;        // 4096 wave tiles
  const int nt = flat & 15;                    // 16 N tiles of 64
  const int mt = flat >> 4;                    // 256 M tiles of 16

  v8f acc[4] = {};
  gemm_mainloop(A + (size_t)mt * 16 * D_, W + (size_t)nt * 64 * D_, lane, acc);

  const int n = lane & 15, base8 = (lane >> 4) * 8;
  #pragma unroll
  for (int c = 0; c < 4; ++c) {
    float bv = bias[nt * 64 + c * 16 + n];
    #pragma unroll
    for (int i = 0; i < 8; ++i) {
      int M = mt * 16 + base8 + i;
      out[(size_t)M * D_ + nt * 64 + c * 16 + n] = acc[c][i] + bv;
    }
  }
}

// ---------------------------------------------------------------- launch
extern "C" void kernel_launch(void* const* d_in, const int* in_sizes, int n_in,
                              void* d_out, int out_size, void* d_ws, size_t ws_size,
                              hipStream_t stream) {
  const float* x     = (const float*)d_in[0];   // (B,S,D)
  const float* w_qkv = (const float*)d_in[1];   // (3D,D)
  const float* w_out = (const float*)d_in[2];   // (D,D)
  const float* b_out = (const float*)d_in[3];   // (D,)
  float* out = (float*)d_out;                   // (B,S,D) f32

  char* ws = (char*)d_ws;
  size_t off = 0;
  _Float16* xf = (_Float16*)(ws + off); off += (size_t)MTOT * D_ * 2;        // 8 MB
  _Float16* wq = (_Float16*)(ws + off); off += (size_t)3 * D_ * D_ * 2;      // 6 MB
  _Float16* wo = (_Float16*)(ws + off); off += (size_t)D_ * D_ * 2;          // 2 MB
  _Float16* qb = (_Float16*)(ws + off); off += (size_t)BH_ * S_ * HD_ * 2;   // 8 MB
  _Float16* kb = (_Float16*)(ws + off); off += (size_t)BH_ * S_ * HD_ * 2;   // 8 MB
  _Float16* vt = (_Float16*)(ws + off); off += (size_t)BH_ * S_ * HD_ * 2;   // 8 MB
  _Float16* ob = (_Float16*)(ws + off); off += (size_t)MTOT * D_ * 2;        // 8 MB

  cvt_f32_f16<<<1024, 256, 0, stream>>>(x,     xf, (MTOT * D_) / 4);
  cvt_f32_f16<<<1024, 256, 0, stream>>>(w_qkv, wq, (3 * D_ * D_) / 4);
  cvt_f32_f16<<<512,  256, 0, stream>>>(w_out, wo, (D_ * D_) / 4);

  qkv_gemm<<<3072, 128, 0, stream>>>(xf, wq, qb, kb, vt);

  const int npairs = BH_ * S_ * (HD_ / 2);      // 2,097,152
  rope_kernel<<<8192, 256, 0, stream>>>(qb, 0.125f, npairs);  // 1/sqrt(64) folded
  rope_kernel<<<8192, 256, 0, stream>>>(kb, 1.0f,   npairs);

  flash_attn<<<1024, 128, 0, stream>>>(qb, kb, vt, ob);

  out_gemm<<<1024, 128, 0, stream>>>(ob, wo, b_out, out);
}
